// TransFusionHead_54855322304577
// MI455X (gfx1250) — matrix operand order repaired and checked by reading, not compiled, not run on hardware
//
#include <hip/hip_runtime.h>
#include <cstddef>

// ---------------------------------------------------------------------------
// CDNA5 (gfx1250) TransFusion head. All heavy GEMM/conv work goes through
// v_wmma_f32_16x16x32_bf16 (wave32, f32 accum). Layouts follow CDNA5 ISA.
// LDS staging of pre-quantized bf16 weights uses async global->LDS (ASYNCcnt)
// when the toolchain exposes the gfx1250 builtin.
// ---------------------------------------------------------------------------

typedef __attribute__((ext_vector_type(16))) __bf16 bf16x16;
typedef __attribute__((ext_vector_type(8)))  float  floatx8;

#define DI __device__ __forceinline__

constexpr int Bc = 2, CINc = 512, Hc = 180, Wcc = 180, HWc = Hc * Wcc;
constexpr int HIDc = 128, NCLSc = 10, NPROPc = 200, FFNd = 256;

// LDS tile strides (elements); multiples of 8 ush = 16B so async b128 stays aligned
constexpr int LDA_A = 40;    // A tile: 128 x 32, stride 40
constexpr int LDB_G = 136;   // gemm B tile: 32 x 128, stride 136
constexpr int LDB_C = 132;   // conv B tile: 32 x 128, stride 132 (ds_store path)

union ABfrag { bf16x16 v; unsigned short h[16]; };

DI unsigned short f2bf(float f) {
  union { float f; unsigned u; } cv; cv.f = f;
  unsigned r = cv.u + 0x7FFFu + ((cv.u >> 16) & 1u);   // round-to-nearest-even
  return (unsigned short)(r >> 16);
}

DI floatx8 zero8() {
  floatx8 z;
  #pragma unroll
  for (int i = 0; i < 8; ++i) z[i] = 0.f;
  return z;
}

DI floatx8 wmma_bf16(const ABfrag& a, const ABfrag& b, floatx8 c) {
  return __builtin_amdgcn_wmma_f32_16x16x32_bf16(false, a.v, false, b.v,
                                                 (short)0, c, false, false);
}

// -------- async global->LDS staging (gfx1250 GLOBAL_LOAD_ASYNC_TO_LDS_B128) ---
#if defined(__gfx1250__) && __has_builtin(__builtin_amdgcn_global_load_async_to_lds_b128)
#define USE_ASYNC_LDS 1
typedef __attribute__((__vector_size__(16))) int v4i_t;
DI void async_copy16B(const void* g, void* l) {   // 16 bytes, ASYNCcnt-tracked
  __builtin_amdgcn_global_load_async_to_lds_b128(
      (__attribute__((address_space(1))) v4i_t*)g,
      (__attribute__((address_space(3))) v4i_t*)l, 0, 0);
}
DI void wait_async0() { asm volatile("s_wait_asynccnt 0" ::: "memory"); }
#else
#define USE_ASYNC_LDS 0
DI void wait_async0() {}
#endif

// A-matrix 16x32 (MxK) bf16 fragment: lanes 0-15 -> M rows, K=0..7,16..23;
// lanes 16-31 -> same M rows, K=8..15,24..31 (ISA 7.12.2).
DI void load_a_frag(ABfrag& f, const unsigned short* As, int row0, int lda, int lane) {
  int lo = lane & 15, hi = lane >> 4;
  const unsigned short* r = As + (row0 + lo) * lda;
  #pragma unroll
  for (int v = 0; v < 8; ++v) {
    int kb = (v < 4 ? 2 * v : 8 + 2 * v) + hi * 8;
    f.h[2 * v]     = r[kb];
    f.h[2 * v + 1] = r[kb + 1];
  }
}

// B-matrix 32x16 (KxN): lanes 0-15 hold K=0..7,16..23; lanes 16-31 hold K=8..15,24..31; N=lane&15.
DI void load_b_frag(ABfrag& f, const unsigned short* Bs, int col0, int ldb, int lane) {
  int lo = lane & 15, hi = lane >> 4;
  const unsigned short* c = Bs + col0 + lo;
  #pragma unroll
  for (int j = 0; j < 16; ++j) f.h[j] = c[((j < 8 ? j : j + 8) + hi * 8) * ldb];
}

// ---------------------------------------------------------------------------
// weight cast / transpose-cast
// ---------------------------------------------------------------------------
__global__ void cast_bf16_k(const float* __restrict__ in, unsigned short* __restrict__ out, int n) {
  int i = blockIdx.x * 256 + threadIdx.x;
  if (i < n) out[i] = f2bf(in[i]);
}

__global__ void transpose_bf16_k(const float* __restrict__ in, unsigned short* __restrict__ out,
                                 int R, int C) {           // in [R,C] -> out [C,R]
  int i = blockIdx.x * 256 + threadIdx.x;
  if (i >= R * C) return;
  int r = i / C, c = i - r * C;
  out[c * R + r] = f2bf(in[i]);
}

// ---------------------------------------------------------------------------
// implicit-GEMM 3x3 conv (pad=1): out[m,pos] = sum_k W[m,k] * patch[k,pos]
// M-tile 128, N-tile 128 positions, K-step 32. 8 waves: 2(M) x 4(N).
// epilogue: (+bias)*gamma+beta (BN, optional) + ReLU.
// ---------------------------------------------------------------------------
__global__ __launch_bounds__(256) void conv3x3_wmma(
    const float* __restrict__ x, int Cin,
    const unsigned short* __restrict__ wq, int Cout,
    const float* __restrict__ bias, const float* __restrict__ gamma,
    const float* __restrict__ beta, int relu, float* __restrict__ out) {
  const int Ktot = Cin * 9;
  const int n0 = blockIdx.x * 128;
  const int bz = blockIdx.z;
  __shared__ unsigned short As[128 * LDA_A];
  __shared__ unsigned short Bs[32 * LDB_C];
  const int tid = threadIdx.x, lane = tid & 31, wid = tid >> 5;
  const int wm = wid >> 2, wn = wid & 3;
  floatx8 acc[4][2];
  #pragma unroll
  for (int i = 0; i < 4; ++i)
    #pragma unroll
    for (int j = 0; j < 2; ++j) acc[i][j] = zero8();

  for (int k0 = 0; k0 < Ktot; k0 += 32) {
    { // stage A (weights, already bf16) -- async global->LDS when available
      int m = tid >> 1, off = (tid & 1) * 16;
      unsigned short* dst = &As[m * LDA_A + off];
      if (m < Cout) {
        const unsigned short* src = wq + (size_t)m * Ktot + k0 + off;
        __builtin_prefetch(src + 32, 0, 0);               // global_prefetch_b8
#if USE_ASYNC_LDS
        async_copy16B(src, dst);
        async_copy16B(src + 8, dst + 8);
#else
        #pragma unroll
        for (int i = 0; i < 16; ++i) dst[i] = src[i];
#endif
      } else {
        #pragma unroll
        for (int i = 0; i < 16; ++i) dst[i] = 0;
      }
    }
    { // stage B (im2col on the fly, f32 -> bf16)
      int k = tid >> 3, nb = (tid & 7) * 16;
      int kk = k0 + k;
      int c = kk / 9, rem = kk - c * 9;
      int dr = rem / 3 - 1, ds = rem - (rem / 3) * 3 - 1;
      const float* xp = x + ((size_t)bz * Cin + c) * HWc;
      #pragma unroll
      for (int i = 0; i < 16; ++i) {
        int pos = n0 + nb + i;
        float val = 0.f;
        if (pos < HWc) {
          int hh = pos / Wcc, w = pos - hh * Wcc;
          int ph = hh + dr, pw = w + ds;
          if ((unsigned)ph < (unsigned)Hc && (unsigned)pw < (unsigned)Wcc)
            val = xp[ph * Wcc + pw];
        }
        Bs[k * LDB_C + nb + i] = f2bf(val);
      }
    }
    wait_async0();
    __syncthreads();
    ABfrag a[4], bf[2];
    #pragma unroll
    for (int i = 0; i < 4; ++i) load_a_frag(a[i], As, wm * 64 + i * 16, LDA_A, lane);
    #pragma unroll
    for (int j = 0; j < 2; ++j) load_b_frag(bf[j], Bs, wn * 32 + j * 16, LDB_C, lane);
    #pragma unroll
    for (int i = 0; i < 4; ++i)
      #pragma unroll
      for (int j = 0; j < 2; ++j) acc[i][j] = wmma_bf16(a[i], bf[j], acc[i][j]);
    __syncthreads();
  }
  // epilogue: D layout -> m = r + 8*(lane>>4), n = lane&15
  int lo = lane & 15, hi = lane >> 4;
  #pragma unroll
  for (int i = 0; i < 4; ++i)
    #pragma unroll
    for (int j = 0; j < 2; ++j)
      #pragma unroll
      for (int r = 0; r < 8; ++r) {
        int m = wm * 64 + i * 16 + r + 8 * hi;
        int n = n0 + wn * 32 + j * 16 + lo;
        if (m < Cout && n < HWc) {
          float v = acc[i][j][r];
          if (gamma) v = (v + bias[m]) * gamma[m] + beta[m];
          else       v = v + bias[m];
          if (relu) v = fmaxf(v, 0.f);
          out[((size_t)bz * Cout + m) * HWc + n] = v;
        }
      }
}

// ---------------------------------------------------------------------------
// generic batched GEMM: C[M,N] = A[M,K](f32,row-major,lda) x B[K,N](bf16,ldb)
// N is always a multiple of 128 here, so the B tile needs no column guard and
// can be staged with async global->LDS b128 copies.
// epilogue: (acc + bias[n]) * scale, optional ReLU; store f32 and/or bf16.
// ---------------------------------------------------------------------------
__global__ __launch_bounds__(256) void gemm_wmma(
    const float* __restrict__ A, long long strideA, int lda,
    const unsigned short* __restrict__ Bm, int ldb,
    float* __restrict__ Cf, long long strideCf,
    unsigned short* __restrict__ Cb, long long strideCb,
    int M, int N, int K,
    const float* __restrict__ bias, int relu, float scale) {
  const int n0 = blockIdx.x * 128, m0 = blockIdx.y * 128, bz = blockIdx.z;
  const float* Ab = A + (size_t)bz * strideA;
  __shared__ unsigned short As[128 * LDA_A];
  __shared__ unsigned short Bs[32 * LDB_G];
  const int tid = threadIdx.x, lane = tid & 31, wid = tid >> 5;
  const int wm = wid >> 2, wn = wid & 3;
  floatx8 acc[4][2];
  #pragma unroll
  for (int i = 0; i < 4; ++i)
    #pragma unroll
    for (int j = 0; j < 2; ++j) acc[i][j] = zero8();

  for (int k0 = 0; k0 < K; k0 += 32) {
    { // B: bf16 weights, contiguous 32B per thread -> async LDS when available
      int k = tid >> 3, nb = (tid & 7) * 16;
      const unsigned short* src = Bm + (size_t)(k0 + k) * ldb + n0 + nb;
      unsigned short* dst = &Bs[k * LDB_G + nb];
#if USE_ASYNC_LDS
      async_copy16B(src, dst);
      async_copy16B(src + 8, dst + 8);
#else
      #pragma unroll
      for (int i = 0; i < 16; ++i) dst[i] = src[i];
#endif
    }
    { // A: f32 -> bf16 (conversion needs VALU; ds_store path)
      int mr = tid >> 1, off = (tid & 1) * 16;
      int mg = m0 + mr;
      unsigned short* dst = &As[mr * LDA_A + off];
      if (mg < M) {
        const float* src = Ab + (size_t)mg * lda + k0 + off;
        #pragma unroll
        for (int i = 0; i < 16; ++i) dst[i] = f2bf(src[i]);
      } else {
        #pragma unroll
        for (int i = 0; i < 16; ++i) dst[i] = 0;
      }
    }
    wait_async0();
    __syncthreads();
    ABfrag a[4], bf[2];
    #pragma unroll
    for (int i = 0; i < 4; ++i) load_a_frag(a[i], As, wm * 64 + i * 16, LDA_A, lane);
    #pragma unroll
    for (int j = 0; j < 2; ++j) load_b_frag(bf[j], Bs, wn * 32 + j * 16, LDB_G, lane);
    #pragma unroll
    for (int i = 0; i < 4; ++i)
      #pragma unroll
      for (int j = 0; j < 2; ++j) acc[i][j] = wmma_bf16(a[i], bf[j], acc[i][j]);
    __syncthreads();
  }
  int lo = lane & 15, hi = lane >> 4;
  #pragma unroll
  for (int i = 0; i < 4; ++i)
    #pragma unroll
    for (int j = 0; j < 2; ++j)
      #pragma unroll
      for (int r = 0; r < 8; ++r) {
        int m = m0 + wm * 64 + i * 16 + r + 8 * hi;
        int n = n0 + wn * 32 + j * 16 + lo;
        if (m < M && n < N) {
          float v = (acc[i][j][r] + bias[n]) * scale;
          if (relu) v = fmaxf(v, 0.f);
          if (Cf) Cf[(size_t)bz * strideCf + (size_t)m * N + n] = v;
          if (Cb) Cb[(size_t)bz * strideCb + (size_t)m * N + n] = f2bf(v);
        }
      }
}

// ---------------------------------------------------------------------------
// sigmoid + 3x3 local-max NMS (sigmoid monotone -> compare raw logits)
// ---------------------------------------------------------------------------
__global__ void nms_sigmoid_k(const float* __restrict__ dense, float* __restrict__ masked) {
  int i = blockIdx.x * 256 + threadIdx.x;
  if (i >= Bc * NCLSc * HWc) return;
  int pos = i % HWc, c = (i / HWc) % NCLSc;
  int hh = pos / Wcc, w = pos - hh * Wcc;
  float v = dense[i];
  float sg = 1.f / (1.f + __expf(-v));
  float o;
  if (c >= 8) o = sg;
  else {
    bool keep = (hh >= 1 && hh <= Hc - 2 && w >= 1 && w <= Wcc - 2);
    if (keep) {
      const float* base = dense + (i - pos);
      float mx = -1e30f;
      #pragma unroll
      for (int dr = -1; dr <= 1; ++dr)
        #pragma unroll
        for (int ds = -1; ds <= 1; ++ds)
          mx = fmaxf(mx, base[(hh + dr) * Wcc + (w + ds)]);
      keep = (v == mx);
    }
    o = keep ? sg : 0.f;
  }
  masked[i] = o;
}

// iterative block argmax top-200 per batch (destroys masked)
__global__ __launch_bounds__(1024) void topk200_k(float* __restrict__ masked, int* __restrict__ top) {
  int b = blockIdx.x, t = threadIdx.x;
  float* base = masked + (size_t)b * NCLSc * HWc;
  __shared__ float bv[1024];
  __shared__ int   bi[1024];
  const int Nt = NCLSc * HWc;
  for (int p = 0; p < NPROPc; ++p) {
    float best = -1e30f; int bidx = 0;
    for (int i = t; i < Nt; i += 1024) {
      float v = base[i];
      if (v > best) { best = v; bidx = i; }
    }
    bv[t] = best; bi[t] = bidx;
    __syncthreads();
    for (int off = 512; off > 0; off >>= 1) {
      if (t < off && bv[t + off] > bv[t]) { bv[t] = bv[t + off]; bi[t] = bi[t + off]; }
      __syncthreads();
    }
    if (t == 0) { top[b * NPROPc + p] = bi[0]; base[bi[0]] = -1e30f; }
    __syncthreads();
  }
}

// gather query features + class embed, query pos, query pos-embed (sp_*)
__global__ void build_queries_k(const int* __restrict__ top, const float* __restrict__ feat,
    const float* __restrict__ ce_w, const float* __restrict__ ce_b,
    const float* __restrict__ w1, const float* __restrict__ b1,
    const float* __restrict__ g, const float* __restrict__ be,
    const float* __restrict__ w2, const float* __restrict__ b2,
    float* __restrict__ q0, float* __restrict__ qpe, float* __restrict__ qpos) {
  int p = blockIdx.x, b = blockIdx.y, t = threadIdx.x;   // 128 threads
  int idx = top[b * NPROPc + p];
  int cls = idx / HWc, pos = idx - cls * HWc;
  int hh = pos / Wcc, w = pos - hh * Wcc;
  float px = w + 0.5f, py = hh + 0.5f;
  size_t row = (size_t)b * NPROPc + p;
  q0[row * HIDc + t] = feat[((size_t)b * HIDc + t) * HWc + pos] + ce_w[t * NCLSc + cls] + ce_b[t];
  __shared__ float h1[HIDc];
  float hv = (px * w1[t * 2] + py * w1[t * 2 + 1] + b1[t]) * g[t] + be[t];
  h1[t] = fmaxf(hv, 0.f);
  __syncthreads();
  float o = b2[t];
  for (int c = 0; c < HIDc; ++c) o += h1[c] * w2[t * HIDc + c];
  qpe[row * HIDc + t] = o;
  if (t == 0) { qpos[row * 2] = px; qpos[row * 2 + 1] = py; }
}

// BEV pos-embed first layer: relu((pos @ w1^T + b1)*g + be) over all HW cells
__global__ void bev_embed_h1_k(const float* __restrict__ w1, const float* __restrict__ b1,
    const float* __restrict__ g, const float* __restrict__ be, float* __restrict__ h1) {
  int i = blockIdx.x * 256 + threadIdx.x;
  if (i >= HWc * HIDc) return;
  int pos = i >> 7, c = i & 127;
  int hh = pos / Wcc, w = pos - hh * Wcc;
  float px = w + 0.5f, py = hh + 0.5f;
  float v = (px * w1[c * 2] + py * w1[c * 2 + 1] + b1[c]) * g[c] + be[c];
  h1[i] = fmaxf(v, 0.f);
}

// kv[b,l,c] = feat[b,c,l] + kpe[l,c]
__global__ void build_kv_k(const float* __restrict__ feat, const float* __restrict__ kpe,
                           float* __restrict__ kv) {
  int i = blockIdx.x * 256 + threadIdx.x;
  if (i >= Bc * HWc * HIDc) return;
  int c = i & 127;
  int l = (i >> 7) % HWc;
  int b = i / (HWc * HIDc);
  kv[i] = feat[((size_t)b * HIDc + c) * HWc + l] + kpe[(size_t)l * HIDc + c];
}

__global__ void add_vec_k(const float* __restrict__ a, const float* __restrict__ b,
                          float* __restrict__ o, int n) {
  int i = blockIdx.x * 256 + threadIdx.x;
  if (i < n) o[i] = a[i] + b[i];
}

// LayerNorm(out = LN(y + resid)) over C=128 rows
__global__ void ln_res_k(const float* __restrict__ y, const float* __restrict__ resid,
                         const float* __restrict__ g, const float* __restrict__ bta,
                         float* __restrict__ out) {
  int r = blockIdx.x, t = threadIdx.x;       // 128 threads
  size_t base = (size_t)r * HIDc;
  float v = y[base + t] + resid[base + t];
  __shared__ float red[HIDc];
  red[t] = v; __syncthreads();
  for (int off = 64; off > 0; off >>= 1) { if (t < off) red[t] += red[t + off]; __syncthreads(); }
  float mean = red[0] * (1.f / HIDc);
  __syncthreads();
  float d = v - mean;
  red[t] = d * d; __syncthreads();
  for (int off = 64; off > 0; off >>= 1) { if (t < off) red[t] += red[t + off]; __syncthreads(); }
  float var = red[0] * (1.f / HIDc);
  out[base + t] = d * rsqrtf(var + 1e-5f) * g[t] + bta[t];
}

// self-attention over 200 queries x 200 keys (scalar; tiny)
__global__ __launch_bounds__(256) void self_attn_k(const float* __restrict__ qkv,
                                                   float* __restrict__ out, float scale) {
  int qi = blockIdx.x, h = blockIdx.y, b = blockIdx.z, t = threadIdx.x;
  const float* base = qkv + (size_t)b * NPROPc * 384;
  const float* qv = base + (size_t)qi * 384 + h * 16;
  __shared__ float sp[NPROPc];
  __shared__ float red[256];
  float s = -1e30f;
  if (t < NPROPc) {
    const float* kv = base + (size_t)t * 384 + 128 + h * 16;
    float d = 0.f;
    #pragma unroll
    for (int c = 0; c < 16; ++c) d += qv[c] * kv[c];
    s = d * scale;
  }
  red[t] = s; __syncthreads();
  for (int off = 128; off > 0; off >>= 1) { if (t < off) red[t] = fmaxf(red[t], red[t + off]); __syncthreads(); }
  float mx = red[0]; __syncthreads();
  float p = (t < NPROPc) ? __expf(s - mx) : 0.f;
  red[t] = p; __syncthreads();
  for (int off = 128; off > 0; off >>= 1) { if (t < off) red[t] += red[t + off]; __syncthreads(); }
  float sum = red[0]; __syncthreads();
  if (t < NPROPc) sp[t] = p / sum;
  __syncthreads();
  if (t < 16) {
    float o = 0.f;
    for (int j = 0; j < NPROPc; ++j) o += sp[j] * base[(size_t)j * 384 + 256 + h * 16 + t];
    out[((size_t)b * NPROPc + qi) * HIDc + h * 16 + t] = o;
  }
}

// ---------------------------------------------------------------------------
// flash cross-attention: 1 wave per (batch, head, 16-query tile);
// streams 32 keys/iter: 2 score WMMAs + online softmax + 1 P@V WMMA.
// ---------------------------------------------------------------------------
__global__ __launch_bounds__(32) void cross_attn_flash_k(
    const float* __restrict__ q,           // [B,200,128], already scaled
    const unsigned short* __restrict__ Kp, // [B,HW,128] bf16
    const unsigned short* __restrict__ Vp, // [B,HW,128] bf16
    float* __restrict__ out) {             // [B,200,128]
  int qt = blockIdx.x, head = blockIdx.y, b = blockIdx.z;
  int lane = threadIdx.x, lo = lane & 15, hi = lane >> 4;
  // Q fragment: A[m=query 0..15][k=hd, padded 16->32 with zeros]
  ABfrag qa;
  int qrow = qt * 16 + lo;
  const float* qp = q + ((size_t)b * NPROPc + (qrow < NPROPc ? qrow : NPROPc - 1)) * HIDc + head * 16;
  #pragma unroll
  for (int v = 0; v < 8; ++v) {
    int kb = (v < 4 ? 2 * v : 8 + 2 * v) + hi * 8;
    qa.h[2 * v]     = (kb     < 16) ? f2bf(qp[kb])     : (unsigned short)0;
    qa.h[2 * v + 1] = (kb + 1 < 16) ? f2bf(qp[kb + 1]) : (unsigned short)0;
  }
  float rowmax[8], rowsum[8];
  #pragma unroll
  for (int r = 0; r < 8; ++r) { rowmax[r] = -1e30f; rowsum[r] = 0.f; }
  floatx8 oacc = zero8();
  __shared__ unsigned short Pl[16 * 32];

  for (int kb0 = 0; kb0 < HWc; kb0 += 32) {
    // K fragments for keys kb0..+15 and kb0+16..+31 (B[k=hd][n=key] = Kp[key][hd])
    ABfrag k1, k2;
    int key1 = kb0 + lo, key2 = kb0 + 16 + lo;
    const unsigned short* kp1 = Kp + ((size_t)b * HWc + (key1 < HWc ? key1 : HWc - 1)) * HIDc + head * 16;
    const unsigned short* kp2 = Kp + ((size_t)b * HWc + (key2 < HWc ? key2 : HWc - 1)) * HIDc + head * 16;
    bool v1 = key1 < HWc, v2 = key2 < HWc;
    #pragma unroll
    for (int j = 0; j < 16; ++j) {
      k1.h[j] = (j < 8 && v1) ? kp1[j + hi * 8] : (unsigned short)0;
      k2.h[j] = (j < 8 && v2) ? kp2[j + hi * 8] : (unsigned short)0;
    }
    floatx8 s1 = wmma_bf16(qa, k1, zero8());
    floatx8 s2 = wmma_bf16(qa, k2, zero8());
    // online softmax; row m = r + 8*hi, cols are lanes (lo) within half
    #pragma unroll
    for (int r = 0; r < 8; ++r) {
      float a1 = v1 ? s1[r] : -1e30f;
      float a2 = v2 ? s2[r] : -1e30f;
      float mx = fmaxf(a1, a2);
      for (int d = 1; d < 16; d <<= 1) mx = fmaxf(mx, __shfl_xor(mx, d, 32));
      float newmax = fmaxf(rowmax[r], mx);
      float corr = __expf(rowmax[r] - newmax);
      float p1 = v1 ? __expf(s1[r] - newmax) : 0.f;
      float p2 = v2 ? __expf(s2[r] - newmax) : 0.f;
      float ps = p1 + p2;
      for (int d = 1; d < 16; d <<= 1) ps += __shfl_xor(ps, d, 32);
      rowsum[r] = rowsum[r] * corr + ps;
      rowmax[r] = newmax;
      oacc[r] *= corr;
      int m = r + 8 * hi;
      Pl[m * 32 + lo]      = f2bf(p1);
      Pl[m * 32 + 16 + lo] = f2bf(p2);
    }
    __syncthreads();
    // P fragment (A[m=query][k=key 0..31]) and V fragment (B[k=key][n=hd])
    ABfrag pa, vf;
    #pragma unroll
    for (int v = 0; v < 8; ++v) {
      int kb = (v < 4 ? 2 * v : 8 + 2 * v) + hi * 8;
      pa.h[2 * v]     = Pl[lo * 32 + kb];
      pa.h[2 * v + 1] = Pl[lo * 32 + kb + 1];
    }
    const unsigned short* vp = Vp + ((size_t)b * HWc) * HIDc + head * 16 + lo;
    #pragma unroll
    for (int j = 0; j < 16; ++j) {
      int key = kb0 + (j < 8 ? j : j + 8) + hi * 8;
      vf.h[j] = (key < HWc) ? vp[(size_t)key * HIDc] : (unsigned short)0;
    }
    oacc = wmma_bf16(pa, vf, oacc);
    __syncthreads();
  }
  #pragma unroll
  for (int r = 0; r < 8; ++r) {
    int qq = qt * 16 + r + 8 * hi;
    if (qq < NPROPc)
      out[((size_t)b * NPROPc + qq) * HIDc + head * 16 + lo] = oacc[r] / rowsum[r];
  }
}

// prediction FFN heads
__global__ void pred_heads_k(const float* __restrict__ q3, const float* __restrict__ qpos,
    const float* __restrict__ pw1, const float* __restrict__ pb1,
    const float* __restrict__ pg1, const float* __restrict__ pbe1,
    const float* __restrict__ wc, const float* __restrict__ bc,
    const float* __restrict__ wh, const float* __restrict__ bh,
    const float* __restrict__ wd, const float* __restrict__ bd,
    const float* __restrict__ wr, const float* __restrict__ br,
    const float* __restrict__ wv, const float* __restrict__ bv,
    const float* __restrict__ whm, const float* __restrict__ bhm,
    float* __restrict__ preds) {
  int p = blockIdx.x, n = blockIdx.y, b = blockIdx.z, t = threadIdx.x;  // 64 thr
  __shared__ float sh[64];
  const float* qrow = q3 + ((size_t)b * NPROPc + p) * HIDc;
  int hidx = n * 64 + t;
  float v = pb1[hidx];
  for (int c = 0; c < HIDc; ++c) v += pw1[(size_t)hidx * HIDc + c] * qrow[c];
  v = v * pg1[hidx] + pbe1[hidx];
  sh[t] = fmaxf(v, 0.f);
  __syncthreads();
  int odim, roff; const float* wp; const float* bp;
  if      (n == 0) { odim = 2;  roff = 0;  wp = wc;  bp = bc;  }
  else if (n == 1) { odim = 1;  roff = 2;  wp = wh;  bp = bh;  }
  else if (n == 2) { odim = 3;  roff = 3;  wp = wd;  bp = bd;  }
  else if (n == 3) { odim = 2;  roff = 6;  wp = wr;  bp = br;  }
  else if (n == 4) { odim = 2;  roff = 8;  wp = wv;  bp = bv;  }
  else             { odim = 10; roff = 10; wp = whm; bp = bhm; }
  if (t < odim) {
    float o = bp[t];
    for (int h = 0; h < 64; ++h) o += wp[t * 64 + h] * sh[h];
    if (n == 0) o += qpos[((size_t)b * NPROPc + p) * 2 + t];
    preds[(size_t)b * 20 * NPROPc + (roff + t) * NPROPc + p] = o;
  }
}

// ---------------------------------------------------------------------------
// host orchestration
// ---------------------------------------------------------------------------
extern "C" void kernel_launch(void* const* d_in, const int* in_sizes, int n_in,
                              void* d_out, int out_size, void* d_ws, size_t ws_size,
                              hipStream_t stream) {
  (void)in_sizes; (void)n_in; (void)out_size; (void)ws_size;
  const float* x      = (const float*)d_in[0];
  const float* sc_w   = (const float*)d_in[1];
  const float* sc_b   = (const float*)d_in[2];
  const float* sc_g   = (const float*)d_in[3];
  const float* sc_be  = (const float*)d_in[4];
  const float* hm_w1  = (const float*)d_in[5];
  const float* hm_b1  = (const float*)d_in[6];
  const float* hm_g1  = (const float*)d_in[7];
  const float* hm_be1 = (const float*)d_in[8];
  const float* hm_w2  = (const float*)d_in[9];
  const float* hm_b2  = (const float*)d_in[10];
  const float* ce_w   = (const float*)d_in[11];
  const float* ce_b   = (const float*)d_in[12];
  const float* sp_w1  = (const float*)d_in[13];
  const float* sp_b1  = (const float*)d_in[14];
  const float* sp_g   = (const float*)d_in[15];
  const float* sp_be  = (const float*)d_in[16];
  const float* sp_w2  = (const float*)d_in[17];
  const float* sp_b2  = (const float*)d_in[18];
  const float* cp_w1  = (const float*)d_in[19];
  const float* cp_b1  = (const float*)d_in[20];
  const float* cp_g   = (const float*)d_in[21];
  const float* cp_be  = (const float*)d_in[22];
  const float* cp_w2  = (const float*)d_in[23];
  const float* cp_b2  = (const float*)d_in[24];
  const float* sa_in_w  = (const float*)d_in[25];
  const float* sa_in_b  = (const float*)d_in[26];
  const float* sa_out_w = (const float*)d_in[27];
  const float* sa_out_b = (const float*)d_in[28];
  const float* ca_in_w  = (const float*)d_in[29];
  const float* ca_in_b  = (const float*)d_in[30];
  const float* ca_out_w = (const float*)d_in[31];
  const float* ca_out_b = (const float*)d_in[32];
  const float* n1_g = (const float*)d_in[33];
  const float* n1_b = (const float*)d_in[34];
  const float* n2_g = (const float*)d_in[35];
  const float* n2_b = (const float*)d_in[36];
  const float* n3_g = (const float*)d_in[37];
  const float* n3_b = (const float*)d_in[38];
  const float* ff_w1 = (const float*)d_in[39];
  const float* ff_b1 = (const float*)d_in[40];
  const float* ff_w2 = (const float*)d_in[41];
  const float* ff_b2 = (const float*)d_in[42];
  const float* pw1  = (const float*)d_in[43];
  const float* pb1  = (const float*)d_in[44];
  const float* pg1  = (const float*)d_in[45];
  const float* pbe1 = (const float*)d_in[46];
  const float* w_center = (const float*)d_in[47];
  const float* b_center = (const float*)d_in[48];
  const float* w_height = (const float*)d_in[49];
  const float* b_height = (const float*)d_in[50];
  const float* w_dim = (const float*)d_in[51];
  const float* b_dim = (const float*)d_in[52];
  const float* w_rot = (const float*)d_in[53];
  const float* b_rot = (const float*)d_in[54];
  const float* w_vel = (const float*)d_in[55];
  const float* b_vel = (const float*)d_in[56];
  const float* w_hm  = (const float*)d_in[57];
  const float* b_hm  = (const float*)d_in[58];

  // workspace arena (buffers reused where lifetimes allow)
  char* wp_ = (char*)d_ws;
  auto alloc = [&](size_t b) -> void* {
    void* p = (void*)wp_; wp_ += (b + 255) & ~(size_t)255; return p;
  };
  float* feat  = (float*)alloc((size_t)Bc * HIDc * HWc * 4);   // conv features
  float* hbuf  = (float*)alloc((size_t)Bc * HIDc * HWc * 4);   // hm mid -> kv
  float* h1bev = (float*)alloc((size_t)HWc * HIDc * 4);        // -> K_proj bf16
  float* kpe   = (float*)alloc((size_t)HWc * HIDc * 4);        // -> V_proj bf16
  float* masked = (float*)alloc((size_t)Bc * NCLSc * HWc * 4);
  int*   topidx = (int*)alloc((size_t)Bc * NPROPc * 4);
  float* q0   = (float*)alloc((size_t)Bc * NPROPc * HIDc * 4);
  float* qpe  = (float*)alloc((size_t)Bc * NPROPc * HIDc * 4);
  float* qpos = (float*)alloc((size_t)Bc * NPROPc * 2 * 4);
  float* tmp  = (float*)alloc((size_t)Bc * NPROPc * HIDc * 4);
  float* qkv  = (float*)alloc((size_t)Bc * NPROPc * 384 * 4);
  float* samid = (float*)alloc((size_t)Bc * NPROPc * HIDc * 4);
  float* ybuf  = (float*)alloc((size_t)Bc * NPROPc * HIDc * 4);
  float* q1 = (float*)alloc((size_t)Bc * NPROPc * HIDc * 4);
  float* q2 = (float*)alloc((size_t)Bc * NPROPc * HIDc * 4);
  float* q3 = (float*)alloc((size_t)Bc * NPROPc * HIDc * 4);
  float* attn  = (float*)alloc((size_t)Bc * NPROPc * HIDc * 4);
  float* ffmid = (float*)alloc((size_t)Bc * NPROPc * FFNd * 4);
  unsigned short* wsc    = (unsigned short*)alloc((size_t)HIDc * CINc * 9 * 2);
  unsigned short* whm1   = (unsigned short*)alloc((size_t)HIDc * HIDc * 9 * 2);
  unsigned short* whm2   = (unsigned short*)alloc((size_t)NCLSc * HIDc * 9 * 2);
  unsigned short* cpw2T  = (unsigned short*)alloc(128 * 128 * 2);
  unsigned short* sainT  = (unsigned short*)alloc(128 * 384 * 2);
  unsigned short* saoutT = (unsigned short*)alloc(128 * 128 * 2);
  unsigned short* cainT  = (unsigned short*)alloc(128 * 384 * 2);
  unsigned short* caoutT = (unsigned short*)alloc(128 * 128 * 2);
  unsigned short* ffw1T  = (unsigned short*)alloc(128 * 256 * 2);
  unsigned short* ffw2T  = (unsigned short*)alloc(256 * 128 * 2);
  unsigned short* Kp = (unsigned short*)h1bev;   // [B,HW,128] bf16 (same bytes)
  unsigned short* Vp = (unsigned short*)kpe;     // [B,HW,128] bf16 (same bytes)

  auto cdiv = [](int a, int b) { return (a + b - 1) / b; };
  const int NT = cdiv(HWc, 128);   // 254 position tiles

  // 1) weight casts / transposed-casts to bf16
  cast_bf16_k<<<cdiv(HIDc * CINc * 9, 256), 256, 0, stream>>>(sc_w, wsc, HIDc * CINc * 9);
  cast_bf16_k<<<cdiv(HIDc * HIDc * 9, 256), 256, 0, stream>>>(hm_w1, whm1, HIDc * HIDc * 9);
  cast_bf16_k<<<cdiv(NCLSc * HIDc * 9, 256), 256, 0, stream>>>(hm_w2, whm2, NCLSc * HIDc * 9);
  transpose_bf16_k<<<cdiv(128 * 128, 256), 256, 0, stream>>>(cp_w2, cpw2T, 128, 128);
  transpose_bf16_k<<<cdiv(384 * 128, 256), 256, 0, stream>>>(sa_in_w, sainT, 384, 128);
  transpose_bf16_k<<<cdiv(128 * 128, 256), 256, 0, stream>>>(sa_out_w, saoutT, 128, 128);
  transpose_bf16_k<<<cdiv(384 * 128, 256), 256, 0, stream>>>(ca_in_w, cainT, 384, 128);
  transpose_bf16_k<<<cdiv(128 * 128, 256), 256, 0, stream>>>(ca_out_w, caoutT, 128, 128);
  transpose_bf16_k<<<cdiv(256 * 128, 256), 256, 0, stream>>>(ff_w1, ffw1T, 256, 128);
  transpose_bf16_k<<<cdiv(128 * 256, 256), 256, 0, stream>>>(ff_w2, ffw2T, 128, 256);

  // 2) convs (implicit GEMM, WMMA)
  conv3x3_wmma<<<dim3(NT, 1, Bc), 256, 0, stream>>>(x, CINc, wsc, HIDc, sc_b, sc_g, sc_be, 1, feat);
  conv3x3_wmma<<<dim3(NT, 1, Bc), 256, 0, stream>>>(feat, HIDc, whm1, HIDc, hm_b1, hm_g1, hm_be1, 1, hbuf);
  float* dense = (float*)d_out + (size_t)Bc * 20 * NPROPc;
  conv3x3_wmma<<<dim3(NT, 1, Bc), 256, 0, stream>>>(hbuf, HIDc, whm2, NCLSc, hm_b2, nullptr, nullptr, 0, dense);

  // 3) NMS + top-k proposals
  nms_sigmoid_k<<<cdiv(Bc * NCLSc * HWc, 256), 256, 0, stream>>>(dense, masked);
  topk200_k<<<Bc, 1024, 0, stream>>>(masked, topidx);

  // 4) BEV key pos-embed + kv + K/V projections (bf16)
  bev_embed_h1_k<<<cdiv(HWc * HIDc, 256), 256, 0, stream>>>(cp_w1, cp_b1, cp_g, cp_be, h1bev);
  gemm_wmma<<<dim3(1, NT, 1), 256, 0, stream>>>(h1bev, 0, 128, cpw2T, 128,
      kpe, 0, nullptr, 0, HWc, 128, 128, cp_b2, 0, 1.f);
  build_kv_k<<<cdiv(Bc * HWc * HIDc, 256), 256, 0, stream>>>(feat, kpe, hbuf);
  gemm_wmma<<<dim3(1, NT, Bc), 256, 0, stream>>>(hbuf, (long long)HWc * 128, 128, cainT + 128, 384,
      nullptr, 0, Kp, (long long)HWc * 128, HWc, 128, 128, ca_in_b + 128, 0, 1.f);
  gemm_wmma<<<dim3(1, NT, Bc), 256, 0, stream>>>(hbuf, (long long)HWc * 128, 128, cainT + 256, 384,
      nullptr, 0, Vp, (long long)HWc * 128, HWc, 128, 128, ca_in_b + 256, 0, 1.f);

  // 5) queries + self-attention block
  build_queries_k<<<dim3(NPROPc, Bc), 128, 0, stream>>>(topidx, feat, ce_w, ce_b,
      sp_w1, sp_b1, sp_g, sp_be, sp_w2, sp_b2, q0, qpe, qpos);
  add_vec_k<<<cdiv(Bc * NPROPc * HIDc, 256), 256, 0, stream>>>(q0, qpe, tmp, Bc * NPROPc * HIDc);
  gemm_wmma<<<dim3(3, 2, Bc), 256, 0, stream>>>(tmp, (long long)NPROPc * 128, 128, sainT, 384,
      qkv, (long long)NPROPc * 384, nullptr, 0, NPROPc, 384, 128, sa_in_b, 0, 1.f);
  self_attn_k<<<dim3(NPROPc, 8, Bc), 256, 0, stream>>>(qkv, samid, 0.25f);
  gemm_wmma<<<dim3(1, 2, Bc), 256, 0, stream>>>(samid, (long long)NPROPc * 128, 128, saoutT, 128,
      ybuf, (long long)NPROPc * 128, nullptr, 0, NPROPc, 128, 128, sa_out_b, 0, 1.f);
  ln_res_k<<<Bc * NPROPc, 128, 0, stream>>>(ybuf, q0, n1_g, n1_b, q1);

  // 6) cross-attention block (flash, WMMA)
  add_vec_k<<<cdiv(Bc * NPROPc * HIDc, 256), 256, 0, stream>>>(q1, qpe, tmp, Bc * NPROPc * HIDc);
  gemm_wmma<<<dim3(1, 2, Bc), 256, 0, stream>>>(tmp, (long long)NPROPc * 128, 128, cainT, 384,
      ybuf, (long long)NPROPc * 128, nullptr, 0, NPROPc, 128, 128, ca_in_b, 0, 0.25f);
  cross_attn_flash_k<<<dim3(13, 8, Bc), 32, 0, stream>>>(ybuf, Kp, Vp, attn);
  gemm_wmma<<<dim3(1, 2, Bc), 256, 0, stream>>>(attn, (long long)NPROPc * 128, 128, caoutT, 128,
      samid, (long long)NPROPc * 128, nullptr, 0, NPROPc, 128, 128, ca_out_b, 0, 1.f);
  ln_res_k<<<Bc * NPROPc, 128, 0, stream>>>(samid, q1, n2_g, n2_b, q2);

  // 7) FFN block
  gemm_wmma<<<dim3(2, 2, Bc), 256, 0, stream>>>(q2, (long long)NPROPc * 128, 128, ffw1T, 256,
      ffmid, (long long)NPROPc * 256, nullptr, 0, NPROPc, 256, 128, ff_b1, 1, 1.f);
  gemm_wmma<<<dim3(1, 2, Bc), 256, 0, stream>>>(ffmid, (long long)NPROPc * 256, 256, ffw2T, 128,
      ybuf, (long long)NPROPc * 128, nullptr, 0, NPROPc, 128, 256, ff_b2, 0, 1.f);
  ln_res_k<<<Bc * NPROPc, 128, 0, stream>>>(ybuf, q2, n3_g, n3_b, q3);

  // 8) prediction heads -> preds [B,20,P] at d_out base
  pred_heads_k<<<dim3(NPROPc, 6, Bc), 64, 0, stream>>>(q3, qpos, pw1, pb1, pg1, pbe1,
      w_center, b_center, w_height, b_height, w_dim, b_dim, w_rot, b_rot,
      w_vel, b_vel, w_hm, b_hm, (float*)d_out);
}
